// MyNeuralSDE_11072425689920
// MI455X (gfx1250) — compile-verified, matrix operand order checked by reading, not run on hardware
//
#include <hip/hip_runtime.h>
#include <hip/hip_bf16.h>

typedef __bf16 bf16;
typedef __attribute__((ext_vector_type(16))) __bf16 v16bf;
typedef __attribute__((ext_vector_type(8)))  __bf16 v8bf;
typedef __attribute__((ext_vector_type(8)))  float  v8f;

#define MT    32      // batch rows per workgroup (2 WMMA M-tiles)
#define ASTR  136     // LDS stride for 128-wide activations (bf16 elems, mult of 8)
#define ISTR  104     // LDS stride for 96-wide (padded 65) input (bf16 elems)
#define NSTEPS 64
#define NTHREADS 256

// d_ws offsets in bf16 elements (fragment-swizzled weights)
#define OFF_WI0 0         // 128 x 32
#define OFF_WI1 4096      // 128 x 128
#define OFF_WI2 20480     // 64  x 128
#define OFF_WV0 28672     // 128 x 96 (padded from 65)
#define OFF_WV1 40960     // 128 x 128
#define OFF_WV2 57344     // 64  x 128
#define OFF_WC0 65536     // 128 x 96 (padded from 65)
#define OFF_WC1 77824     // 128 x 128
#define OFF_WC2 94208     // 2048 x 128

// ---- gfx1250 hardware tanh (V_TANH_F32) with portable fallback ----
#if __has_builtin(__builtin_amdgcn_tanhf)
#define FAST_TANH(x) __builtin_amdgcn_tanhf(x)
#else
#define FAST_TANH(x) tanhf(x)
#endif

// ---- Tensor Data Mover availability (arity differs across toolchains) ----
#if __has_builtin(__builtin_amdgcn_tensor_load_to_lds)
#define HAVE_TDM 1
typedef unsigned int u32x4 __attribute__((ext_vector_type(4)));
typedef int          i32x4 __attribute__((ext_vector_type(4)));
typedef int          i32x8 __attribute__((ext_vector_type(8)));
#endif

// ---------------------------------------------------------------------------
// Prep: f32 row-major weights [N,Kreal] -> bf16 fragment-swizzled layout
// [nt][ks][lane][16] so each lane's B fragment is one coalesced 32B load.
//   row = nt*16 + (L&15)
//   col = ks*32 + ((L>>4)<<3) + e + (e>=8 ? 8 : 0)   (zero-pad col >= Kreal)
// ---------------------------------------------------------------------------
__global__ void prep_frag_kernel(const float* __restrict__ src, bf16* __restrict__ dst,
                                 int Kreal, int Kpad, int total) {
  int idx = blockIdx.x * blockDim.x + threadIdx.x;
  if (idx >= total) return;
  int KS = Kpad >> 5;
  int e  = idx & 15;
  int L  = (idx >> 4) & 31;
  int fs = idx >> 9;
  int ks = fs % KS;
  int nt = fs / KS;
  int row = nt * 16 + (L & 15);
  int col = ks * 32 + ((L >> 4) << 3) + e + ((e >= 8) ? 8 : 0);
  float v = (col < Kreal) ? src[(size_t)row * Kreal + col] : 0.f;
  dst[idx] = (bf16)v;
}

// ---------------------------------------------------------------------------
// Device helpers
// ---------------------------------------------------------------------------
__device__ inline v8f vzero8() { v8f z = {}; return z; }

// Load one A fragment (16x32 bf16) from LDS row-major activations.
__device__ inline v16bf lda(const bf16* __restrict__ A, int astr, int ks, int lane) {
  const bf16* p = A + (lane & 15) * astr + ks * 32 + ((lane >> 4) << 3);
  v8bf lo = *(const v8bf*)p;
  v8bf hi = *(const v8bf*)(p + 16);
  v16bf r;
#pragma unroll
  for (int i = 0; i < 8; ++i) { r[i] = lo[i]; r[8 + i] = hi[i]; }
  return r;
}

template<int KS>
__device__ inline void load_afrags(const bf16* __restrict__ A0, const bf16* __restrict__ A1,
                                   int astr, int lane, v16bf af0[KS], v16bf af1[KS]) {
#pragma unroll
  for (int ks = 0; ks < KS; ++ks) {
    af0[ks] = lda(A0, astr, ks, lane);
    af1[ks] = lda(A1, astr, ks, lane);
  }
}

// One N-tile GEMM over both M-tiles using preloaded A fragments.
template<int KS>
__device__ inline void gemm2f(const v16bf af0[KS], const v16bf af1[KS],
                              const bf16* __restrict__ Wf, int nt, int lane,
                              v8f& acc0, v8f& acc1) {
#pragma unroll
  for (int ks = 0; ks < KS; ++ks) {
    v16bf b = *(const v16bf*)(Wf + ((((nt * KS) + ks) * 32 + lane) << 4));
    acc0 = __builtin_amdgcn_wmma_f32_16x16x32_bf16(false, af0[ks], false, b, (short)0, acc0, false, false);
    acc1 = __builtin_amdgcn_wmma_f32_16x16x32_bf16(false, af1[ks], false, b, (short)0, acc1, false, false);
  }
}

// Epilogue: bias + activation, store bf16 to LDS [row][col].
// ACT: 0 = relu, 1 = lipswish (0.909*x*sigmoid(x) = 0.4545*x*(1+tanh(x/2)))
template<int ACT>
__device__ inline void store_act(bf16* __restrict__ out, int ostr, int col, int lane,
                                 const v8f& acc, float bias) {
  int hi8 = (lane >> 4) << 3;
#pragma unroll
  for (int r = 0; r < 8; ++r) {
    float x = acc[r] + bias;
    float f;
    if (ACT == 0) f = fmaxf(x, 0.f);
    else          f = 0.4545f * x * (1.f + FAST_TANH(0.5f * x));
    out[(r + hi8) * ostr + col] = (bf16)f;
  }
}

struct Params {
  const float *ts, *noise, *bm;
  const float *bi0, *bi1, *bi2;
  const float *bv0, *bv1, *bv2;
  const float *bc0, *bc1, *bc2;
  const float *scale_v, *scale_c, *Wr, *br;
  const bf16* wf;
  float* out;
};

// ---------------------------------------------------------------------------
// Main per-tile SDE kernel: each block carries MT batch rows through the full
// 64-step scan. No inter-block communication.
// ---------------------------------------------------------------------------
__global__ __launch_bounds__(NTHREADS) void sde_scan_kernel(Params p) {
  __shared__ __align__(32) bf16 inp[MT][ISTR];     // [t, y] padded to 96 cols
  __shared__ __align__(32) bf16 act[4][MT][ASTR];  // h1v, h1c, h2v, h2c
  __shared__ float ybuf[MT][64];
  __shared__ __align__(16) float bmbuf[MT][32];
  __shared__ float wrbuf[8 * 64];
  __shared__ float brbuf[8];

  const int tid  = threadIdx.x;
  const int wave = tid >> 5;
  const int lane = tid & 31;
  const int hi8  = (lane >> 4) << 3;
  const int b0   = blockIdx.x * MT;

  const bf16* Wi0 = p.wf + OFF_WI0;
  const bf16* Wi1 = p.wf + OFF_WI1;
  const bf16* Wi2 = p.wf + OFF_WI2;
  const bf16* Wv0 = p.wf + OFF_WV0;
  const bf16* Wv1 = p.wf + OFF_WV1;
  const bf16* Wv2 = p.wf + OFF_WV2;
  const bf16* Wc0 = p.wf + OFF_WC0;
  const bf16* Wc1 = p.wf + OFF_WC1;
  const bf16* Wc2 = p.wf + OFF_WC2;

  // ---- one-time init: zero inp (incl. K padding), stage readout weights ----
  for (int k = tid; k < MT * ISTR; k += NTHREADS) (&inp[0][0])[k] = (bf16)0.f;
  for (int k = tid; k < 512; k += NTHREADS) wrbuf[k] = p.Wr[k];
  if (tid < 8) brbuf[tid] = p.br[tid];
  for (int k = tid; k < MT * 32; k += NTHREADS) {   // init_noise -> inp cols 0..31
    int b = k >> 5, n = k & 31;
    inp[b][n] = (bf16)p.noise[(size_t)(b0 + b) * 32 + n];
  }
  __syncthreads();

  // ---- initial MLP: noise -> y0 (relu hidden, identity final) ----
  {  // L0i
    v16bf a0[1], a1[1];
    load_afrags<1>(&inp[0][0], &inp[16][0], ISTR, lane, a0, a1);
    v8f d0 = vzero8(), d1 = vzero8();
    gemm2f<1>(a0, a1, Wi0, wave, lane, d0, d1);
    int col = wave * 16 + (lane & 15);
    float bias = p.bi0[col];
    store_act<0>(&act[0][0][0],  ASTR, col, lane, d0, bias);
    store_act<0>(&act[0][16][0], ASTR, col, lane, d1, bias);
  }
  __syncthreads();
  {  // L1i
    v16bf a0[4], a1[4];
    load_afrags<4>(&act[0][0][0], &act[0][16][0], ASTR, lane, a0, a1);
    v8f d0 = vzero8(), d1 = vzero8();
    gemm2f<4>(a0, a1, Wi1, wave, lane, d0, d1);
    int col = wave * 16 + (lane & 15);
    float bias = p.bi1[col];
    store_act<0>(&act[1][0][0],  ASTR, col, lane, d0, bias);
    store_act<0>(&act[1][16][0], ASTR, col, lane, d1, bias);
  }
  __syncthreads();
  if (wave < 4) {  // L2i: identity -> y0
    v16bf a0[4], a1[4];
    load_afrags<4>(&act[1][0][0], &act[1][16][0], ASTR, lane, a0, a1);
    v8f d0 = vzero8(), d1 = vzero8();
    gemm2f<4>(a0, a1, Wi2, wave, lane, d0, d1);
    int col = wave * 16 + (lane & 15);
    float bias = p.bi2[col];
#pragma unroll
    for (int r = 0; r < 8; ++r) {
      ybuf[r + hi8][col]      = d0[r] + bias;
      ybuf[16 + r + hi8][col] = d1[r] + bias;
    }
  }
  __syncthreads();

  // ---- 64-step scan (dt = 1, sqrt_dt = 1) ----
  const float t0 = p.ts[0];
  for (int i = 0; i < NSTEPS; ++i) {
#if defined(HAVE_TDM)
    // Stage bm[b0:b0+MT, i, :] (32x32 f32, row stride 2048 elems) via the
    // Tensor Data Mover; wave 0 issues, completes before the barrier.
    if (wave == 0) {
      unsigned long long ga = (unsigned long long)(uintptr_t)p.bm +
                              (((unsigned long long)b0 * NSTEPS + (unsigned)i) * 32ull) * 4ull;
      unsigned int lds_off = (unsigned int)(uintptr_t)(void*)&bmbuf[0][0];
      u32x4 g0;
      g0[0] = 1u;                                        // count=1 valid descriptor
      g0[1] = lds_off;                                   // lds_addr (bytes)
      g0[2] = (unsigned int)ga;                          // global_addr[31:0]
      g0[3] = (unsigned int)(ga >> 32) | (2u << 30);     // global_addr[56:32] | type=2
      i32x8 g1;
      g1[0] = 0x00020000;        // workgroup_mask=0, data_size=2 (4B)
      g1[1] = (int)(32u << 16);  // tensor_dim0 = 32 (bits[79:48], low half)
      g1[2] = (int)(32u << 16);  // tensor_dim0 hi=0 | tensor_dim1 = 32 (bits[111:80])
      g1[3] = (int)(32u << 16);  // tensor_dim1 hi=0 | tile_dim0 = 32 (bits[127:112])
      g1[4] = 32;                // tile_dim1 = 32, tile_dim2 = 0
      g1[5] = 2048;              // tensor_dim0_stride = STEPS*NZ = 2048 elems
      g1[6] = 0;
      g1[7] = 0;
      i32x4 gz = {};             // 2D tensor: groups 2/3 unused
#if __clang_major__ >= 23
      i32x8 gz8 = {};
      __builtin_amdgcn_tensor_load_to_lds(g0, g1, gz, gz, gz8, 0);
#else
      __builtin_amdgcn_tensor_load_to_lds(g0, g1, gz, gz, 0);
#endif
    }
#else
    for (int k = tid; k < MT * 32; k += NTHREADS) {
      int b = k >> 5, n = k & 31;
      bmbuf[b][n] = p.bm[((size_t)(b0 + b) * NSTEPS + i) * 32 + n];
    }
#endif
    // build inp = [t, y] (cols 65..103 stay zero)
    float tcur = t0 + (float)i;
    for (int k = tid; k < MT * 64; k += NTHREADS) {
      int b = k >> 6, h = k & 63;
      if (h == 0) inp[b][0] = (bf16)tcur;
      inp[b][h + 1] = (bf16)ybuf[b][h];
    }
#if defined(HAVE_TDM)
    if (wave == 0) __builtin_amdgcn_s_wait_tensorcnt(0);
#endif
    __syncthreads();

    // L0 (K padded to 96): inp x Wv0 -> act0 ; inp x Wc0 -> act1 (lipswish)
    {
      v16bf a0[3], a1[3];
      load_afrags<3>(&inp[0][0], &inp[16][0], ISTR, lane, a0, a1);  // shared A
      int col = wave * 16 + (lane & 15);
      v8f d0 = vzero8(), d1 = vzero8();
      gemm2f<3>(a0, a1, Wv0, wave, lane, d0, d1);
      float bias = p.bv0[col];
      store_act<1>(&act[0][0][0],  ASTR, col, lane, d0, bias);
      store_act<1>(&act[0][16][0], ASTR, col, lane, d1, bias);
      v8f c0 = vzero8(), c1 = vzero8();
      gemm2f<3>(a0, a1, Wc0, wave, lane, c0, c1);
      bias = p.bc0[col];
      store_act<1>(&act[1][0][0],  ASTR, col, lane, c0, bias);
      store_act<1>(&act[1][16][0], ASTR, col, lane, c1, bias);
    }
    __syncthreads();

    // L1: act0 x Wv1 -> act2 ; act1 x Wc1 -> act3 (lipswish)
    {
      int col = wave * 16 + (lane & 15);
      v16bf a0[4], a1[4];
      load_afrags<4>(&act[0][0][0], &act[0][16][0], ASTR, lane, a0, a1);
      v8f d0 = vzero8(), d1 = vzero8();
      gemm2f<4>(a0, a1, Wv1, wave, lane, d0, d1);
      float bias = p.bv1[col];
      store_act<1>(&act[2][0][0],  ASTR, col, lane, d0, bias);
      store_act<1>(&act[2][16][0], ASTR, col, lane, d1, bias);
      load_afrags<4>(&act[1][0][0], &act[1][16][0], ASTR, lane, a0, a1);
      v8f c0 = vzero8(), c1 = vzero8();
      gemm2f<4>(a0, a1, Wc1, wave, lane, c0, c1);
      bias = p.bc1[col];
      store_act<1>(&act[3][0][0],  ASTR, col, lane, c0, bias);
      store_act<1>(&act[3][16][0], ASTR, col, lane, c1, bias);
    }
    __syncthreads();

    // L2v (waves 0..3): drift = scale_v * tanh(h2v @ Wv2^T + bv2); y += drift
    if (wave < 4) {
      v16bf a0[4], a1[4];
      load_afrags<4>(&act[2][0][0], &act[2][16][0], ASTR, lane, a0, a1);
      v8f d0 = vzero8(), d1 = vzero8();
      gemm2f<4>(a0, a1, Wv2, wave, lane, d0, d1);
      int col = wave * 16 + (lane & 15);
      float bias = p.bv2[col];
      float sv = p.scale_v[col];
#pragma unroll
      for (int r = 0; r < 8; ++r) {
        ybuf[r + hi8][col]      += sv * FAST_TANH(d0[r] + bias);
        ybuf[16 + r + hi8][col] += sv * FAST_TANH(d1[r] + bias);
      }
    }
    __syncthreads();

    // L2c + fused einsum: wave w owns N-tiles [16w,16w+16) -> h in [8w,8w+8)
    // diff[b, h*32+n] = scale_c * tanh(h2c @ Wc2^T + bc2); y[b,h] += sum_n diff*bm
    {
      v16bf a0[4], a1[4];  // A fragments invariant across all 16 N-tiles
      load_afrags<4>(&act[3][0][0], &act[3][16][0], ASTR, lane, a0, a1);
#pragma unroll 2
      for (int j = 0; j < 16; ++j) {
        int nt = wave * 16 + j;
        if (j + 1 < 16)  // pull next tile's fragments into cache (global_prefetch_b8)
          __builtin_prefetch(Wc2 + ((((nt + 1) * 4) * 32) << 4), 0, 1);
        v8f c0 = vzero8(), c1 = vzero8();
        gemm2f<4>(a0, a1, Wc2, nt, lane, c0, c1);
        int col = nt * 16 + (lane & 15);   // 0..2047
        int h = col >> 5, n = col & 31;    // h uniform across the tile
        float bias = p.bc2[col];
        float sc = p.scale_c[col];
        float q0[8], q1[8];
#pragma unroll
        for (int r = 0; r < 8; ++r) {
          q0[r] = FAST_TANH(c0[r] + bias) * sc * bmbuf[r + hi8][n];
          q1[r] = FAST_TANH(c1[r] + bias) * sc * bmbuf[16 + r + hi8][n];
        }
#pragma unroll
        for (int m = 1; m < 16; m <<= 1) {
#pragma unroll
          for (int r = 0; r < 8; ++r) {
            q0[r] += __shfl_xor(q0[r], m, 32);
            q1[r] += __shfl_xor(q1[r], m, 32);
          }
        }
        if ((lane & 15) == 0) {  // lanes 0 & 16: rows 0-7 / 8-15 of each M-tile
#pragma unroll
          for (int r = 0; r < 8; ++r) {
            ybuf[r + hi8][h]      += q0[r];
            ybuf[16 + r + hi8][h] += q1[r];
          }
        }
      }
    }
    __syncthreads();

    // Readout: out[b, i, d] = y[b,:] . Wr[d,:] + br[d]   (256 threads = 32x8)
    {
      int b = tid >> 3, d = tid & 7;
      float s = brbuf[d];
#pragma unroll 8
      for (int h = 0; h < 64; ++h) s += ybuf[b][h] * wrbuf[d * 64 + h];
      p.out[((size_t)(b0 + b) * NSTEPS + i) * 8 + d] = s;
    }
    __syncthreads();
  }
}

// ---------------------------------------------------------------------------
extern "C" void kernel_launch(void* const* d_in, const int* in_sizes, int n_in,
                              void* d_out, int out_size, void* d_ws, size_t ws_size,
                              hipStream_t stream) {
  const float* ts    = (const float*)d_in[0];
  const float* noise = (const float*)d_in[1];
  const float* bm    = (const float*)d_in[2];
  const float* Wi0 = (const float*)d_in[3];  const float* bi0 = (const float*)d_in[4];
  const float* Wi1 = (const float*)d_in[5];  const float* bi1 = (const float*)d_in[6];
  const float* Wi2 = (const float*)d_in[7];  const float* bi2 = (const float*)d_in[8];
  const float* scale_v = (const float*)d_in[9];
  const float* Wv0 = (const float*)d_in[10]; const float* bv0 = (const float*)d_in[11];
  const float* Wv1 = (const float*)d_in[12]; const float* bv1 = (const float*)d_in[13];
  const float* Wv2 = (const float*)d_in[14]; const float* bv2 = (const float*)d_in[15];
  const float* scale_c = (const float*)d_in[16];
  const float* Wc0 = (const float*)d_in[17]; const float* bc0 = (const float*)d_in[18];
  const float* Wc1 = (const float*)d_in[19]; const float* bc1 = (const float*)d_in[20];
  const float* Wc2 = (const float*)d_in[21]; const float* bc2 = (const float*)d_in[22];
  const float* Wr  = (const float*)d_in[23]; const float* br  = (const float*)d_in[24];

  bf16* wf = (bf16*)d_ws;

  struct Job { const float* src; int off, N, Kreal, Kpad; };
  const Job jobs[9] = {
    {Wi0, OFF_WI0, 128,  32,  32}, {Wi1, OFF_WI1, 128, 128, 128},
    {Wi2, OFF_WI2,  64, 128, 128}, {Wv0, OFF_WV0, 128,  65,  96},
    {Wv1, OFF_WV1, 128, 128, 128}, {Wv2, OFF_WV2,  64, 128, 128},
    {Wc0, OFF_WC0, 128,  65,  96}, {Wc1, OFF_WC1, 128, 128, 128},
    {Wc2, OFF_WC2, 2048, 128, 128},
  };
  for (int j = 0; j < 9; ++j) {
    int total = jobs[j].N * jobs[j].Kpad;
    int grid = (total + 255) / 256;
    prep_frag_kernel<<<grid, 256, 0, stream>>>(jobs[j].src, wf + jobs[j].off,
                                               jobs[j].Kreal, jobs[j].Kpad, total);
  }

  Params p;
  p.ts = ts; p.noise = noise; p.bm = bm;
  p.bi0 = bi0; p.bi1 = bi1; p.bi2 = bi2;
  p.bv0 = bv0; p.bv1 = bv1; p.bv2 = bv2;
  p.bc0 = bc0; p.bc1 = bc1; p.bc2 = bc2;
  p.scale_v = scale_v; p.scale_c = scale_c; p.Wr = Wr; p.br = br;
  p.wf = wf; p.out = (float*)d_out;

  sde_scan_kernel<<<2048 / MT, NTHREADS, 0, stream>>>(p);
}
